// RGCN_17119739642414
// MI455X (gfx1250) — matrix operand order repaired and checked by reading, not compiled, not run on hardware
//
#include <hip/hip_runtime.h>

typedef __attribute__((ext_vector_type(16))) __bf16    v16bf;
typedef __attribute__((ext_vector_type(8)))  float     v8f;
typedef __attribute__((ext_vector_type(8)))  unsigned  v8u;

constexpr int NN  = 100000;   // nodes
constexpr int EE  = 400000;   // edges per relation
constexpr int RR  = 4;        // relations

// ---- bf16 packing helpers ----------------------------------------------
// Fast path (A operand): round-to-nearest (+0x8000), pack 2 f32 -> 1 dword
// of bf16 pairs with a single v_perm_b32.
__device__ __forceinline__ unsigned pack2_rn(float lo, float hi) {
  unsigned ul = __builtin_bit_cast(unsigned, lo) + 0x8000u;
  unsigned uh = __builtin_bit_cast(unsigned, hi) + 0x8000u;
  // out = { uh[31:16], ul[31:16] } : sel bytes {7,6} from src0=uh, {3,2} from src1=ul
  return __builtin_amdgcn_perm(uh, ul, 0x07060302u);
}
// Precise path (W pre-pack, runs once): full round-to-nearest-even.
__device__ __forceinline__ unsigned pack2_rne(float lo, float hi) {
  unsigned ul = __builtin_bit_cast(unsigned, lo);
  unsigned uh = __builtin_bit_cast(unsigned, hi);
  ul += 0x7FFFu + ((ul >> 16) & 1u);
  uh += 0x7FFFu + ((uh >> 16) & 1u);
  return __builtin_amdgcn_perm(uh, ul, 0x07060302u);
}

__global__ void k_zero(float* __restrict__ p, int n) {
  int i = blockIdx.x * blockDim.x + threadIdx.x;
  if (i < n) p[i] = 0.0f;
}

__global__ void k_degrees(const int* __restrict__ src, const int* __restrict__ dst,
                          float* __restrict__ dgo, float* __restrict__ dgi, int total) {
  int i = blockIdx.x * blockDim.x + threadIdx.x;
  if (i >= total) return;
  int r = i / EE;
  unsafeAtomicAdd(&dgo[r * NN + src[i]], 1.0f);
  unsafeAtomicAdd(&dgi[r * NN + dst[i]], 1.0f);
}

__global__ void k_rsqrt(float* __restrict__ p, int n) {
  int i = blockIdx.x * blockDim.x + threadIdx.x;
  if (i >= n) return;
  float d = p[i];
  p[i] = d > 0.0f ? rsqrtf(d) : 1.0f;   // clip(deg,1) then rsqrt
}

// Pre-pack W[din=128 x DOUT] (f32 row-major) into bf16 B-fragment layout:
//   dword index = ((kc*NT + j)*32 + lane)*8 + d
//   lane = hi*16+lid holds col N=lid, K = kc*32 + hi*16 + 2d .. +1
template <int NT>
__global__ void k_packW(const float* __restrict__ W, unsigned* __restrict__ Bp) {
  constexpr int DOUT = NT * 16;
  constexpr int TOTAL = 4 * NT * 32 * 8;
  int i = blockIdx.x * blockDim.x + threadIdx.x;
  if (i >= TOTAL) return;
  int d    = i & 7;
  int lane = (i >> 3) & 31;
  int f    = i >> 8;                 // kc*NT + j
  int kc = f / NT, j = f - kc * NT;
  int lid = lane & 15, hi = lane >> 4;
  int k = kc * 32 + hi * 16 + 2 * d;
  float lo  = W[(size_t)k       * DOUT + j * 16 + lid];
  float hif = W[(size_t)(k + 1) * DOUT + j * 16 + lid];
  Bp[i] = pack2_rne(lo, hif);
}

// hn = (h * rsqrt_deg_out) @ W ; K=128, one wave = 16 rows x (NT*16) cols.
// B comes pre-packed in fragment layout -> contiguous b128 loads, no convert.
template <int NT>
__global__ __launch_bounds__(128) void k_gemm(const float* __restrict__ h,
                                              const unsigned* __restrict__ Bp,
                                              const float* __restrict__ rs,
                                              float* __restrict__ hn, int nrowblocks) {
  constexpr int DOUT = NT * 16;
  int wid = blockIdx.x * 4 + (threadIdx.x >> 5);
  if (wid >= nrowblocks) return;                 // uniform per wave
  int lane = threadIdx.x & 31;
  int lid = lane & 15, hi = lane >> 4;
  int arow = wid * 16 + lid;
  float scale = rs[arow];
  const float* ap = h + (size_t)arow * 128;
  const v16bf* bp = (const v16bf*)Bp;

  v8f acc[NT] = {};
#pragma unroll
  for (int kc = 0; kc < 4; ++kc) {
    const int k0 = kc * 32;
    // A fragment: lanes 0-15 rows M=lid K {0..15}; lanes 16-31 K {16..31}
    v8u ad;
#pragma unroll
    for (int t = 0; t < 4; ++t) {
      ad[t]     = pack2_rn(ap[k0 + hi * 16 + 2 * t] * scale,
                           ap[k0 + hi * 16 + 2 * t + 1] * scale);
      ad[t + 4] = pack2_rn(ap[k0 + 8 + hi * 16 + 2 * t] * scale,
                           ap[k0 + 8 + hi * 16 + 2 * t + 1] * scale);
    }
    v16bf a = __builtin_bit_cast(v16bf, ad);
#pragma unroll
    for (int j = 0; j < NT; ++j) {
      v16bf b = bp[(kc * NT + j) * 32 + lane];   // 32B contiguous per lane
      acc[j] = __builtin_amdgcn_wmma_f32_16x16x32_bf16(
          false, a, false, b, (short)0, acc[j], false, false);
    }
  }
  // D: VGPR v -> row (hi*8+v), col lid
#pragma unroll
  for (int j = 0; j < NT; ++j)
#pragma unroll
    for (int v = 0; v < 8; ++v)
      hn[(size_t)(wid * 16 + hi * 8 + v) * DOUT + j * 16 + lid] = acc[j][v];
}

// agg[dst] += hn[src]  (float4 gather, 4 fire-and-forget f32 atomics; L2-resident)
template <int DOUT>
__global__ void k_scatter(const float* __restrict__ hn, const int* __restrict__ src,
                          const int* __restrict__ dst, float* __restrict__ agg, int total4) {
  int i = blockIdx.x * blockDim.x + threadIdx.x;
  if (i >= total4) return;
  constexpr int G = DOUT / 4;
  int e = i / G;
  int j = (i - e * G) * 4;
  int s = src[e], d = dst[e];
  const float4 v = *reinterpret_cast<const float4*>(hn + (size_t)s * DOUT + j);
  float* o = agg + (size_t)d * DOUT + j;
  unsafeAtomicAdd(o + 0, v.x);
  unsafeAtomicAdd(o + 1, v.y);
  unsafeAtomicAdd(o + 2, v.z);
  unsafeAtomicAdd(o + 3, v.w);
}

// out += relu(agg * rsqrt_deg_in + b) / R
template <int DOUT, bool RELU>
__global__ void k_finalize(const float* __restrict__ agg, const float* __restrict__ rin,
                           const float* __restrict__ b, float* __restrict__ out, int total) {
  int i = blockIdx.x * blockDim.x + threadIdx.x;
  if (i >= total) return;
  int n = i / DOUT;
  int j = i - n * DOUT;
  float v = agg[i] * rin[n] + b[j];
  if (RELU) v = fmaxf(v, 0.0f);
  out[i] += v * 0.25f;   // 1/R
}

static inline int nblk(long n, int b) { return (int)((n + b - 1) / b); }

extern "C" void kernel_launch(void* const* d_in, const int* in_sizes, int n_in,
                              void* d_out, int out_size, void* d_ws, size_t ws_size,
                              hipStream_t stream) {
  const float* x  = (const float*)d_in[0];
  const float* W0 = (const float*)d_in[1];
  const float* b0 = (const float*)d_in[2];
  const float* W1 = (const float*)d_in[3];
  const float* b1 = (const float*)d_in[4];
  const float* W2 = (const float*)d_in[5];
  const float* b2 = (const float*)d_in[6];
  const int*   src = (const int*)d_in[7];
  const int*   dst = (const int*)d_in[8];
  float* out = (float*)d_out;

  float* ws   = (float*)d_ws;
  float* dgo  = ws;                              // [R*N] -> rsqrt(deg_out)
  float* dgi  = dgo + (size_t)RR * NN;           // [R*N] -> rsqrt(deg_in)
  float* bufA = dgi + (size_t)RR * NN;           // [N*128] h after layer0
  float* bufB = bufA + (size_t)NN * 128;         // [N*128] h after layer1
  float* hn   = bufB + (size_t)NN * 128;         // [N*128] GEMM output (per relation)
  float* agg  = hn + (size_t)NN * 128;           // [N*128] scatter target (per relation)
  unsigned* Bp = (unsigned*)(agg + (size_t)NN * 128);  // [8192] packed bf16 W fragments

  const int B = 256;
  const int nrb = NN / 16;                       // 6250 row blocks
  const int gemmBlocks = (nrb + 3) / 4;          // 4 waves / block

  // ---- degrees (shared by all layers) ----
  k_zero<<<nblk(2L * RR * NN, B), B, 0, stream>>>(dgo, 2 * RR * NN);
  k_degrees<<<nblk((long)RR * EE, B), B, 0, stream>>>(src, dst, dgo, dgi, RR * EE);
  k_rsqrt<<<nblk(2L * RR * NN, B), B, 0, stream>>>(dgo, 2 * RR * NN);

  // ---- layer 0: x[N,128] -> bufA[N,128], relu ----
  k_zero<<<nblk((long)NN * 128, B), B, 0, stream>>>(bufA, NN * 128);
  for (int r = 0; r < RR; ++r) {
    k_packW<8><<<nblk(4 * 8 * 32 * 8, B), B, 0, stream>>>(W0 + (size_t)r * 128 * 128, Bp);
    k_gemm<8><<<gemmBlocks, 128, 0, stream>>>(x, Bp, dgo + (size_t)r * NN, hn, nrb);
    k_zero<<<nblk((long)NN * 128, B), B, 0, stream>>>(agg, NN * 128);
    k_scatter<128><<<nblk((long)EE * 32, B), B, 0, stream>>>(hn, src + (size_t)r * EE, dst + (size_t)r * EE, agg, EE * 32);
    k_finalize<128, true><<<nblk((long)NN * 128, B), B, 0, stream>>>(agg, dgi + (size_t)r * NN, b0 + (size_t)r * 128, bufA, NN * 128);
  }

  // ---- layer 1: bufA -> bufB, relu ----
  k_zero<<<nblk((long)NN * 128, B), B, 0, stream>>>(bufB, NN * 128);
  for (int r = 0; r < RR; ++r) {
    k_packW<8><<<nblk(4 * 8 * 32 * 8, B), B, 0, stream>>>(W1 + (size_t)r * 128 * 128, Bp);
    k_gemm<8><<<gemmBlocks, 128, 0, stream>>>(bufA, Bp, dgo + (size_t)r * NN, hn, nrb);
    k_zero<<<nblk((long)NN * 128, B), B, 0, stream>>>(agg, NN * 128);
    k_scatter<128><<<nblk((long)EE * 32, B), B, 0, stream>>>(hn, src + (size_t)r * EE, dst + (size_t)r * EE, agg, EE * 32);
    k_finalize<128, true><<<nblk((long)NN * 128, B), B, 0, stream>>>(agg, dgi + (size_t)r * NN, b1 + (size_t)r * 128, bufB, NN * 128);
  }

  // ---- layer 2: bufB -> out[N,16], no relu ----
  k_zero<<<nblk((long)NN * 16, B), B, 0, stream>>>(out, NN * 16);
  for (int r = 0; r < RR; ++r) {
    k_packW<1><<<nblk(4 * 1 * 32 * 8, B), B, 0, stream>>>(W2 + (size_t)r * 128 * 16, Bp);
    k_gemm<1><<<gemmBlocks, 128, 0, stream>>>(bufB, Bp, dgo + (size_t)r * NN, hn, nrb);
    k_zero<<<nblk((long)NN * 16, B), B, 0, stream>>>(agg, NN * 16);
    k_scatter<16><<<nblk((long)EE * 4, B), B, 0, stream>>>(hn, src + (size_t)r * EE, dst + (size_t)r * EE, agg, EE * 4);
    k_finalize<16, false><<<nblk((long)NN * 16, B), B, 0, stream>>>(agg, dgi + (size_t)r * NN, b2 + (size_t)r * 16, out, NN * 16);
  }
}
